// S4DLin_36361193128109
// MI455X (gfx1250) — matrix operand order repaired and checked by reading, not compiled, not run on hardware
//
#include <hip/hip_runtime.h>
#include <hip/hip_bf16.h>
#include <stdint.h>

// ---------------- problem constants (match reference) ----------------
#define BSZ     4
#define LSEQ    4096
#define DIMSZ   1024
#define DINNER  1024
#define DSTATE  16
#define MROWS   (BSZ * LSEQ)      // 16384
#define LC      512               // chunk length for the scan
#define NCH     (LSEQ / LC)       // 8 chunks

// GEMM tiling
#define MBLK    128
#define NBLK    128
#define KBLK    32
#define ROWP    40                // padded LDS row stride (elements) -> 80B

typedef float  v8f   __attribute__((ext_vector_type(8)));
typedef float  v4f   __attribute__((ext_vector_type(4)));
typedef __bf16 v16bf __attribute__((ext_vector_type(16)));
typedef __bf16 v8bf  __attribute__((ext_vector_type(8)));
typedef __bf16 v4bf  __attribute__((ext_vector_type(4)));

#if defined(__AMDGCN__) && __has_builtin(__builtin_amdgcn_global_load_async_to_lds_b128)
#define HAVE_ASYNC_LDS 1
#else
#define HAVE_ASYNC_LDS 0
#endif

#if HAVE_ASYNC_LDS
// Builtin prototype (from clang diagnostic): param0 is a non-const pointer to
// a 16B int vector in the global address space; param1 the LDS counterpart.
typedef int b128i __attribute__((vector_size(16)));
typedef __attribute__((address_space(1))) b128i* g_b128p;
typedef __attribute__((address_space(3))) b128i* l_b128p;
#endif

// 16B global -> LDS copy. Async (ASYNCcnt-tracked) when available.
__device__ __forceinline__ void copy16_g2l(const __bf16* g, __bf16* l) {
#if HAVE_ASYNC_LDS
    g_b128p gp = (g_b128p)(uintptr_t)g;
    l_b128p lp = (l_b128p)(uint32_t)(uintptr_t)l;   // low 32 bits = LDS offset
    __builtin_amdgcn_global_load_async_to_lds_b128(gp, lp, 0, 0);
#else
    *(v8bf*)l = *(const v8bf*)g;
#endif
}

__device__ __forceinline__ void async_wait0() {
#if HAVE_ASYNC_LDS
#if __has_builtin(__builtin_amdgcn_s_wait_asynccnt)
    __builtin_amdgcn_s_wait_asynccnt(0);
#else
    asm volatile("s_wait_asynccnt 0x0" ::: "memory");
#endif
#endif
}

// ---------------------------------------------------------------------
// One-time fp32 -> (hi, lo) bf16 split (bf16x3 precision recovery).
// ---------------------------------------------------------------------
__global__ void split_hilo(const float* __restrict__ in,
                           __bf16* __restrict__ hi,
                           __bf16* __restrict__ lo, int n) {
    int i = (blockIdx.x * blockDim.x + threadIdx.x) * 4;
    if (i >= n) return;
    v4f x = *(const v4f*)(in + i);
    v4bf h, l;
#pragma unroll
    for (int j = 0; j < 4; ++j) {
        __bf16 hh = (__bf16)x[j];
        h[j] = hh;
        l[j] = (__bf16)(x[j] - (float)hh);
    }
    *(v4bf*)(hi + i) = h;
    *(v4bf*)(lo + i) = l;
}

// ---------------------------------------------------------------------
// NT GEMM: C[M,N] = A[M,K] * Bw[N,K]^T, fp32-accurate via bf16x3 WMMA.
// 128x128 block tile staged through LDS with double-buffered async
// global->LDS copies; 8 waves (2m x 4n), each computing 64x32 via
// v_wmma_f32_16x16x32_bf16 (6 WMMA per m-subtile per 32-k block).
// Fragment layouts used here (wave32; A and B share a contiguous k order):
//   A (16x32 bf16): m = lane&15, elem p -> k = p + 16*(lane>>4)
//   B (32x16 bf16): n = lane&15, elem p -> k = p + 16*(lane>>4)
//   C/D (16x16 f32): n = lane&15, vgpr q -> m = q + 8*(lane>>4)
// ---------------------------------------------------------------------
__global__ __launch_bounds__(256)
void s4d_gemm_bf16x3(const __bf16* __restrict__ Ah, const __bf16* __restrict__ Al,
                     const __bf16* __restrict__ Bh, const __bf16* __restrict__ Bl,
                     float* __restrict__ C, int M, int N, int K) {
    __shared__ __align__(16) __bf16 sA[2][2][MBLK * ROWP];   // [buf][hi/lo]
    __shared__ __align__(16) __bf16 sB[2][2][NBLK * ROWP];

    const int tid  = threadIdx.x;
    const int lane = tid & 31;
    const int wave = tid >> 5;
    const int nblk = N / NBLK;
    const int m0b  = (blockIdx.x / nblk) * MBLK;
    const int n0b  = (blockIdx.x % nblk) * NBLK;

    const int wm = (wave >> 2) * 64;      // wave m-offset within block
    const int wn = (wave & 3) * 32;       // wave n-offset within block
    const int ml = lane & 15;
    const int h  = lane >> 4;

    const int nkb = K / KBLK;
    v8f acc[4][2] = {};

    // Stage one 32-k slab of all four operand tiles into LDS buffer `buf`.
    // 512 x 16B transfers per tile; 2 per thread per tile.
    auto stage = [&](int kb, int buf) {
        const int k0 = kb * KBLK;
#pragma unroll
        for (int it = 0; it < 2; ++it) {
            const int idx = tid + it * 256;           // 0..511
            const int row = idx >> 2;
            const int seg = (idx & 3) * 8;            // element offset in row
            const size_t ga = (size_t)(m0b + row) * K + k0 + seg;
            const size_t gb = (size_t)(n0b + row) * K + k0 + seg;
            const int    la = row * ROWP + seg;
            copy16_g2l(Ah + ga, &sA[buf][0][la]);
            copy16_g2l(Al + ga, &sA[buf][1][la]);
            copy16_g2l(Bh + gb, &sB[buf][0][la]);
            copy16_g2l(Bl + gb, &sB[buf][1][la]);
        }
    };

    stage(0, 0);
    async_wait0();
    __syncthreads();

    for (int kb = 0; kb < nkb; ++kb) {
        const int buf = kb & 1;
        if (kb + 1 < nkb) stage(kb + 1, buf ^ 1);

        // B fragments for the two 16-wide n-subtiles
        const int rb0 = (wn + ml) * ROWP + 16 * h;
        const int rb1 = (wn + 16 + ml) * ROWP + 16 * h;
        union { v16bf v; v8bf p[2]; } b0h, b0l, b1h, b1l;
        b0h.p[0] = *(const v8bf*)&sB[buf][0][rb0];
        b0h.p[1] = *(const v8bf*)&sB[buf][0][rb0 + 8];
        b0l.p[0] = *(const v8bf*)&sB[buf][1][rb0];
        b0l.p[1] = *(const v8bf*)&sB[buf][1][rb0 + 8];
        b1h.p[0] = *(const v8bf*)&sB[buf][0][rb1];
        b1h.p[1] = *(const v8bf*)&sB[buf][0][rb1 + 8];
        b1l.p[0] = *(const v8bf*)&sB[buf][1][rb1];
        b1l.p[1] = *(const v8bf*)&sB[buf][1][rb1 + 8];

#pragma unroll
        for (int t = 0; t < 4; ++t) {
            const int raw = (wm + t * 16 + ml) * ROWP + 16 * h;
            union { v16bf v; v8bf p[2]; } a_hi, a_lo;
            a_hi.p[0] = *(const v8bf*)&sA[buf][0][raw];
            a_hi.p[1] = *(const v8bf*)&sA[buf][0][raw + 8];
            a_lo.p[0] = *(const v8bf*)&sA[buf][1][raw];
            a_lo.p[1] = *(const v8bf*)&sA[buf][1][raw + 8];

            acc[t][0] = __builtin_amdgcn_wmma_f32_16x16x32_bf16(
                            false, a_hi.v, false, b0h.v, (short)0, acc[t][0], false, false);
            acc[t][0] = __builtin_amdgcn_wmma_f32_16x16x32_bf16(
                            false, a_hi.v, false, b0l.v, (short)0, acc[t][0], false, false);
            acc[t][0] = __builtin_amdgcn_wmma_f32_16x16x32_bf16(
                            false, a_lo.v, false, b0h.v, (short)0, acc[t][0], false, false);
            acc[t][1] = __builtin_amdgcn_wmma_f32_16x16x32_bf16(
                            false, a_hi.v, false, b1h.v, (short)0, acc[t][1], false, false);
            acc[t][1] = __builtin_amdgcn_wmma_f32_16x16x32_bf16(
                            false, a_hi.v, false, b1l.v, (short)0, acc[t][1], false, false);
            acc[t][1] = __builtin_amdgcn_wmma_f32_16x16x32_bf16(
                            false, a_lo.v, false, b1h.v, (short)0, acc[t][1], false, false);
        }

        if (kb + 1 < nkb) async_wait0();
        __syncthreads();
    }

#pragma unroll
    for (int t = 0; t < 4; ++t) {
#pragma unroll
        for (int j = 0; j < 2; ++j) {
#pragma unroll
            for (int q = 0; q < 8; ++q) {
                C[(size_t)(m0b + wm + t * 16 + q + 8 * h) * N + n0b + wn + 16 * j + ml]
                    = acc[t][j][q];
            }
        }
    }
}

// ---------------------------------------------------------------------
// Per-(e,s) S4D discretization: Abar = exp(dt*A), coef = C * (Abar-1)/A * B,
// plus Abar^LC (closed form) for cross-chunk composition.
// P layout: [e*16+s][8] = {a_re, a_im, c_re, c_im, aL_re, aL_im, 0, 0}
// ---------------------------------------------------------------------
__global__ void s4d_params(const float* __restrict__ A_log,
                           const float* __restrict__ A_im,
                           const float* __restrict__ B_proj,
                           const float* __restrict__ C_proj,
                           const float* __restrict__ dt_log,
                           float* __restrict__ P) {
    int i = blockIdx.x * blockDim.x + threadIdx.x;
    if (i >= DINNER * DSTATE) return;
    int e = i >> 4;

    float dt   = log1pf(expf(dt_log[e]));          // softplus
    float a_re = -expf(A_log[i]);
    float a_im = A_im[i];
    float lre = dt * a_re, lim = dt * a_im;        // log(Abar)
    float er = expf(lre);
    float Ar = er * cosf(lim);
    float Ai = er * sinf(lim);
    // (Abar - 1) / A
    float nr = Ar - 1.0f, ni = Ai;
    float inv = 1.0f / (a_re * a_re + a_im * a_im);
    float qr = (nr * a_re + ni * a_im) * inv;
    float qi = (ni * a_re - nr * a_im) * inv;
    float bb = B_proj[i];
    float cc = C_proj[i];
    float cr = cc * qr * bb;
    float ci = cc * qi * bb;
    // Abar^LC = exp(LC * log(Abar))
    float prl = expf(lre * (float)LC);
    float aLr = prl * cosf(lim * (float)LC);
    float aLi = prl * sinf(lim * (float)LC);

    float* o = P + (size_t)i * 8;
    o[0] = Ar;  o[1] = Ai;
    o[2] = cr;  o[3] = ci;
    o[4] = aLr; o[5] = aLi;
    o[6] = 0.0f; o[7] = 0.0f;
}

// ---------------------------------------------------------------------
// Scan pass 1: per (b, chunk, e) run the 16-state complex recurrence from
// zero init, store chunk-end states S[b][c][e][s][{re,im}].
// ---------------------------------------------------------------------
__global__ __launch_bounds__(256)
void s4d_scan1(const float* __restrict__ u,
               const float* __restrict__ P,
               float* __restrict__ S) {
    int t = blockIdx.x * blockDim.x + threadIdx.x;   // [b][c][e]
    int e  = t % DINNER;
    int bc = t / DINNER;
    int c  = bc % NCH;
    int b  = bc / NCH;
    if (b >= BSZ) return;

    float ar[DSTATE], ai[DSTATE], hr[DSTATE], hi[DSTATE];
    const float* p = P + (size_t)e * DSTATE * 8;
#pragma unroll
    for (int s = 0; s < DSTATE; ++s) {
        ar[s] = p[s * 8 + 0];
        ai[s] = p[s * 8 + 1];
        hr[s] = 0.0f;
        hi[s] = 0.0f;
    }

    const float* up = u + ((size_t)b * LSEQ + (size_t)c * LC) * DINNER + e;
    for (int j = 0; j < LC; ++j) {
        __builtin_prefetch(up + (size_t)(j + 8) * DINNER, 0, 1);
        float uv = up[(size_t)j * DINNER];
#pragma unroll
        for (int s = 0; s < DSTATE; ++s) {
            float nr = fmaf(ar[s], hr[s], fmaf(-ai[s], hi[s], uv));
            float ni = fmaf(ar[s], hi[s], ai[s] * hr[s]);
            hr[s] = nr;
            hi[s] = ni;
        }
    }

    float* o = S + ((size_t)(b * NCH + c) * DINNER + e) * (2 * DSTATE);
#pragma unroll
    for (int s = 0; s < DSTATE; ++s) {
        o[2 * s + 0] = hr[s];
        o[2 * s + 1] = hi[s];
    }
}

// ---------------------------------------------------------------------
// Scan pass 2: compose incoming state G = sum_k aL^(c-1-k) * S[b][k],
// rerun the chunk, and emit y = 2*Re(sum_s c_s h_s) + D*u directly as
// hi/lo bf16 (fused split for the output GEMM).
// ---------------------------------------------------------------------
__global__ __launch_bounds__(256)
void s4d_scan2(const float* __restrict__ u,
               const float* __restrict__ P,
               const float* __restrict__ S,
               const float* __restrict__ D_skip,
               __bf16* __restrict__ yh,
               __bf16* __restrict__ yl) {
    int t = blockIdx.x * blockDim.x + threadIdx.x;   // [b][c][e]
    int e  = t % DINNER;
    int bc = t / DINNER;
    int c  = bc % NCH;
    int b  = bc / NCH;
    if (b >= BSZ) return;

    float ar[DSTATE], ai[DSTATE], cr[DSTATE], ci[DSTATE];
    float hr[DSTATE], hi[DSTATE];
    const float* p = P + (size_t)e * DSTATE * 8;
#pragma unroll
    for (int s = 0; s < DSTATE; ++s) {
        ar[s] = p[s * 8 + 0];
        ai[s] = p[s * 8 + 1];
        cr[s] = p[s * 8 + 2];
        ci[s] = p[s * 8 + 3];
        hr[s] = 0.0f;
        hi[s] = 0.0f;
    }

    // incoming state from previous chunks: G = aL*G + S_k, k = 0..c-1
    for (int k = 0; k < c; ++k) {
        const float* sk = S + ((size_t)(b * NCH + k) * DINNER + e) * (2 * DSTATE);
#pragma unroll
        for (int s = 0; s < DSTATE; ++s) {
            float aLr = p[s * 8 + 4];
            float aLi = p[s * 8 + 5];
            float nr = fmaf(aLr, hr[s], fmaf(-aLi, hi[s], sk[2 * s + 0]));
            float ni = fmaf(aLr, hi[s], fmaf( aLi, hr[s], sk[2 * s + 1]));
            hr[s] = nr;
            hi[s] = ni;
        }
    }

    float dsk = D_skip[e];
    const size_t base = ((size_t)b * LSEQ + (size_t)c * LC) * DINNER + e;
    const float* up = u + base;
    for (int j = 0; j < LC; ++j) {
        __builtin_prefetch(up + (size_t)(j + 8) * DINNER, 0, 1);
        float uv  = up[(size_t)j * DINNER];
        float acc = 0.0f;
#pragma unroll
        for (int s = 0; s < DSTATE; ++s) {
            float nr = fmaf(ar[s], hr[s], fmaf(-ai[s], hi[s], uv));
            float ni = fmaf(ar[s], hi[s], ai[s] * hr[s]);
            hr[s] = nr;
            hi[s] = ni;
            acc = fmaf(cr[s], nr, fmaf(-ci[s], ni, acc));
        }
        float yv = fmaf(dsk, uv, 2.0f * acc);
        __bf16 hh = (__bf16)yv;
        yh[base + (size_t)j * DINNER] = hh;
        yl[base + (size_t)j * DINNER] = (__bf16)(yv - (float)hh);
    }
}

// ---------------------------------------------------------------------
extern "C" void kernel_launch(void* const* d_in, const int* in_sizes, int n_in,
                              void* d_out, int out_size, void* d_ws, size_t ws_size,
                              hipStream_t stream) {
    const float* x      = (const float*)d_in[0];
    const float* in_w   = (const float*)d_in[1];
    const float* out_w  = (const float*)d_in[2];
    const float* A_log  = (const float*)d_in[3];
    const float* A_im   = (const float*)d_in[4];
    const float* B_proj = (const float*)d_in[5];
    const float* C_proj = (const float*)d_in[6];
    const float* dt_log = (const float*)d_in[7];
    const float* D_skip = (const float*)d_in[8];
    float* out = (float*)d_out;

    char* ws = (char*)d_ws;
    const size_t MB = 1024 * 1024;
    // [0,32MB)    : x_hi, reused as y_hi
    // [32,64MB)   : x_lo, reused as y_lo
    // [64,128MB)  : u (f32)
    // [128,136MB) : weight splits (w1h, w1l, w2h, w2l; 2MB each)
    // [136,137MB) : P params
    // [137,141MB) : S chunk states
    __bf16* Ahb = (__bf16*)(ws);
    __bf16* Alb = (__bf16*)(ws + 32 * MB);
    float*  u   = (float*)(ws + 64 * MB);
    __bf16* w1h = (__bf16*)(ws + 128 * MB);
    __bf16* w1l = (__bf16*)(ws + 130 * MB);
    __bf16* w2h = (__bf16*)(ws + 132 * MB);
    __bf16* w2l = (__bf16*)(ws + 134 * MB);
    float*  P   = (float*)(ws + 136 * MB);
    float*  S   = (float*)(ws + 137 * MB);

    const int nX = MROWS * DINNER;      // 16.7M
    const int nW = DINNER * DIMSZ;      // 1M

    // one-time hi/lo bf16 splits (memory-bound, L2-resident afterwards)
    split_hilo<<<(nX / 4 + 255) / 256, 256, 0, stream>>>(x, Ahb, Alb, nX);
    split_hilo<<<(nW / 4 + 255) / 256, 256, 0, stream>>>(in_w, w1h, w1l, nW);
    split_hilo<<<(nW / 4 + 255) / 256, 256, 0, stream>>>(out_w, w2h, w2l, nW);

    s4d_params<<<(DINNER * DSTATE + 255) / 256, 256, 0, stream>>>(
        A_log, A_im, B_proj, C_proj, dt_log, P);

    // u = x @ in_w^T   (M=16384, N=1024, K=1024)
    {
        int blocks = (MROWS / MBLK) * (DINNER / NBLK);   // 1024
        s4d_gemm_bf16x3<<<blocks, 256, 0, stream>>>(
            Ahb, Alb, w1h, w1l, u, MROWS, DINNER, DIMSZ);
    }

    // chunked complex-diagonal scan; pass 2 writes y as hi/lo bf16 into the
    // (now free) x-split buffers.
    {
        int threads = BSZ * NCH * DINNER;            // 32768
        s4d_scan1<<<threads / 256, 256, 0, stream>>>(u, P, S);
        s4d_scan2<<<threads / 256, 256, 0, stream>>>(u, P, S, D_skip, Ahb, Alb);
    }

    // out = y @ out_w^T  (M=16384, N=1024, K=1024)
    {
        int blocks = (MROWS / MBLK) * (DIMSZ / NBLK);    // 1024
        s4d_gemm_bf16x3<<<blocks, 256, 0, stream>>>(
            Ahb, Alb, w2h, w2l, out, MROWS, DIMSZ, DINNER);
    }
}
